// VoxelHashTableDynamic_738734375106
// MI455X (gfx1250) — matrix operand order repaired and checked by reading, not compiled, not run on hardware
//
#include <hip/hip_runtime.h>

// ---- vector types -----------------------------------------------------------
typedef _Float16 v16h __attribute__((ext_vector_type(16)));
typedef _Float16 v8h  __attribute__((ext_vector_type(8)));
typedef _Float16 v4h  __attribute__((ext_vector_type(4)));
typedef float    v8f  __attribute__((ext_vector_type(8)));

#define DPAD      128           // padded D (K and N)
#define NMAT      8             // td{q,k,v,o}, sd{q,k,v,o}
#define WTS_ELEMS (NMAT * DPAD * DPAD)

// ============================================================================
// Weight prep: f32 [120x120] -> f16 [128x128], n-major (Wt[n][k]),
// head-permuted so head h owns columns [16h,16h+15) with lane 15 zero-padded.
// q/k/v: permute OUTPUT columns.  o: permute INPUT rows (K).
// ============================================================================
__global__ void vox_prep_weights(const float* __restrict__ tdw, const float* __restrict__ tdb,
                                 const float* __restrict__ sdw, const float* __restrict__ sdb,
                                 _Float16* __restrict__ wts, float* __restrict__ bias)
{
    int gid = blockIdx.x * blockDim.x + threadIdx.x;
    if (gid < WTS_ELEMS) {
        int mat = gid >> 14;          // 0..7
        int rem = gid & 16383;
        int n   = rem >> 7;           // padded output col (n-major layout)
        int k   = rem & 127;          // padded K
        int t   = mat & 3;            // 0=q 1=k 2=v 3=o
        const float* W = ((mat < 4) ? tdw : sdw) + (size_t)t * 14400;
        float val = 0.0f;
        if (t < 3) {
            if ((n & 15) < 15 && k < 120)
                val = W[k * 120 + (n >> 4) * 15 + (n & 15)];
        } else {
            if ((k & 15) < 15 && n < 120)
                val = W[((k >> 4) * 15 + (k & 15)) * 120 + n];
        }
        wts[gid] = (_Float16)val;
    }
    if (gid < NMAT * DPAD) {
        int mat = gid >> 7;
        int n   = gid & 127;
        int t   = mat & 3;
        const float* B = ((mat < 4) ? tdb : sdb) + t * 120;
        float val = 0.0f;
        if (t < 3) { if ((n & 15) < 15) val = B[(n >> 4) * 15 + (n & 15)]; }
        else       { if (n < 120)       val = B[n]; }
        bias[gid] = val;
    }
}

// ============================================================================
// 16x(8 n-tiles of 16) GEMM tile: acc = A[16x128] @ Wt^T-tile + bias(col)
// A: LDS, row-major [16][128] f16.  Wt: global, n-major [128][128] f16.
// CDNA5 16-bit A layout: lane l -> M=l&15, K in {g*8..g*8+7, 16+g*8..+7}, g=l>>4
// CDNA5 16-bit B layout: lane l -> N=l&15, K in {g*8..g*8+7, 16+g*8..+7}, g=l>>4
// ============================================================================
__device__ __forceinline__ v8f gemm_tile(const _Float16* A, const _Float16* Wt,
                                         const float* bias, int lane, int tile)
{
    int nn = tile * 16 + (lane & 15);
    float bv = bias[nn];
    v8f acc = { bv, bv, bv, bv, bv, bv, bv, bv };
    int m   = lane & 15;
    int g   = (lane >> 4) & 1;
    const _Float16* arow = A  + m  * DPAD + g * 8;
    const _Float16* brow = Wt + nn * DPAD + g * 8;
#pragma unroll
    for (int kk = 0; kk < 4; ++kk) {
        int k0 = kk * 32;
        v8h alo = *(const v8h*)(arow + k0);
        v8h ahi = *(const v8h*)(arow + k0 + 16);
        v16h a  = __builtin_shufflevector(alo, ahi, 0,1,2,3,4,5,6,7,8,9,10,11,12,13,14,15);
        v8h blo = *(const v8h*)(brow + k0);
        v8h bhi = *(const v8h*)(brow + k0 + 16);
        v16h b  = __builtin_shufflevector(blo, bhi, 0,1,2,3,4,5,6,7,8,9,10,11,12,13,14,15);
        acc = __builtin_amdgcn_wmma_f32_16x16x32_f16(false, a, false, b, (short)0, acc,
                                                     false, false);
    }
    return acc;
}

// f32 row (120 floats) -> f16 LDS row; each half-wave lane covers 60 cols.
__device__ __forceinline__ void gather_row(_Float16* dst, const float* src, int half)
{
#pragma unroll
    for (int i = 0; i < 15; ++i) {
        int c = half * 60 + i * 4;
        float4 v = *(const float4*)(src + c);
        v4h hv = { (_Float16)v.x, (_Float16)v.y, (_Float16)v.z, (_Float16)v.w };
        *(v4h*)(dst + c) = hv;
    }
}

// ============================================================================
// One 2-token attention fusion over this wave's 16 rows.
// A = token0 features, B = token1 features, O = attention output (scratch).
// finalStore=false: write fused token back into B (f16).
// finalStore=true : masked f32 store to global out.
// ============================================================================
__device__ __forceinline__ void fusion_pass(
    _Float16 (&A)[16][DPAD], _Float16 (&B)[16][DPAD], _Float16 (&O)[16][DPAD],
    const _Float16* Wq, const _Float16* Wk, const _Float16* Wv, const _Float16* Wo,
    const float* bq, const float* bk, const float* bv, const float* bo,
    int lane, bool finalStore, float* out, int rowBase, const int* validArr, int M)
{
    const float scale = 0.2581988897471611f;   // 1/sqrt(15)
    int g = (lane >> 4) & 1;
#pragma unroll 1
    for (int tile = 0; tile < 8; ++tile) {     // tile == head (head-padded weights)
        v8f qa = gemm_tile(&A[0][0], Wq, bq, lane, tile);
        v8f ka = gemm_tile(&A[0][0], Wk, bk, lane, tile);
        v8f kb = gemm_tile(&B[0][0], Wk, bk, lane, tile);
        v8f va = gemm_tile(&A[0][0], Wv, bv, lane, tile);
        v8f vb = gemm_tile(&B[0][0], Wv, bv, lane, tile);
        int nn = tile * 16 + (lane & 15);
#pragma unroll
        for (int j = 0; j < 8; ++j) {
            float t0 = qa[j] * ka[j];
            float t1 = qa[j] * kb[j];
#pragma unroll
            for (int msk = 1; msk < 16; msk <<= 1) {   // 16-lane butterfly = head dot
                t0 += __shfl_xor(t0, msk, 32);
                t1 += __shfl_xor(t1, msk, 32);
            }
            t0 *= scale; t1 *= scale;
            float mx = fmaxf(t0, t1);
            float e0 = __expf(t0 - mx);
            float e1 = __expf(t1 - mx);
            float o  = (e0 * va[j] + e1 * vb[j]) / (e0 + e1);
            O[j + g * 8][nn] = (_Float16)o;
        }
    }
    __syncthreads();
#pragma unroll 1
    for (int tile = 0; tile < 8; ++tile) {
        v8f acc = gemm_tile(&O[0][0], Wo, bo, lane, tile);
        int nn = tile * 16 + (lane & 15);
        if (!finalStore) {
#pragma unroll
            for (int j = 0; j < 8; ++j)
                B[j + g * 8][nn] = (_Float16)acc[j];
        } else {
#pragma unroll
            for (int j = 0; j < 8; ++j) {
                int m  = j + g * 8;
                int gr = rowBase + m;
                if (nn < 120 && gr < M)
                    out[(size_t)gr * 120 + nn] = validArr[m] ? acc[j] : 0.0f;
            }
        }
    }
}

// ============================================================================
// Fused kernel: hash -> gather -> fusion(df,te) -> fusion(sf,cond) -> store
// 128 threads = 4 wave32; each wave owns 16 rows; block owns 64 rows.
// ============================================================================
__global__ __launch_bounds__(128) void vox_fused_kernel(
    const float* __restrict__ pts, const int* __restrict__ times,
    const int* __restrict__ buf,
    const float* __restrict__ sf, const float* __restrict__ df,
    const float* __restrict__ te,
    const _Float16* __restrict__ wts, const float* __restrict__ bias,
    float* __restrict__ out, int M)
{
    __shared__ __align__(16) _Float16 sA[4][16][DPAD];
    __shared__ __align__(16) _Float16 sB[4][16][DPAD];
    __shared__ __align__(16) _Float16 sO[4][16][DPAD];
    __shared__ int sValid[4][16];

    int tid  = threadIdx.x;
    int lane = tid & 31;
    int w    = tid >> 5;
    int half = (lane >> 4) & 1;
    int rrow = lane & 15;
    int rowBase = blockIdx.x * 64 + w * 16;
    int grow = rowBase + rrow;

    // --- voxel hash lookup (both half-lanes redundantly compute same row) ---
    int vi = 0, ti = 0, valid = 0;
    if (grow < M) {
        float px = pts[grow * 3 + 0];
        float py = pts[grow * 3 + 1];
        float pz = pts[grow * 3 + 2];
        long long gx = (long long)floorf(px / 0.1f);
        long long gy = (long long)floorf(py / 0.1f);
        long long gz = (long long)floorf(pz / 0.1f);
        long long hs = gx * 73856093ll + gy * 19349669ll + gz * 83492791ll;
        int h = (int)(hs & (long long)((1 << 20) - 1));   // == nonneg mod 2^20
        int v = buf[h];
        valid = (v >= 0) ? 1 : 0;
        vi = valid ? v : 0;
        ti = times[grow] % 201;
    }
    if (half == 0) sValid[w][rrow] = valid;

    // --- gather dynamic features & time embeddings, f32 -> f16, K-pad 0 ----
    gather_row(&sA[w][rrow][0], df + (size_t)vi * 120, half);
    gather_row(&sB[w][rrow][0], te + (size_t)ti * 120, half);
    if (half == 0) {
        v8h z = {};
        *(v8h*)&sA[w][rrow][120] = z;
        *(v8h*)&sB[w][rrow][120] = z;
    }
    __syncthreads();

    const _Float16* Wtd = wts;
    const _Float16* Wsd = wts + 4 * DPAD * DPAD;
    const float* btd = bias;
    const float* bsd = bias + 4 * DPAD;

    // fusion 1: cond = attn(df, te) -> overwrites sB (te dead)
    fusion_pass(sA[w], sB[w], sO[w],
                Wtd + 0 * DPAD * DPAD, Wtd + 1 * DPAD * DPAD,
                Wtd + 2 * DPAD * DPAD, Wtd + 3 * DPAD * DPAD,
                btd + 0 * DPAD, btd + 1 * DPAD, btd + 2 * DPAD, btd + 3 * DPAD,
                lane, false, nullptr, rowBase, &sValid[w][0], M);
    __syncthreads();

    // gather static features into A (df dead); K-pad already zero
    gather_row(&sA[w][rrow][0], sf + (size_t)vi * 120, half);
    __syncthreads();

    // fusion 2: fused = attn(sf, cond) -> masked f32 store
    fusion_pass(sA[w], sB[w], sO[w],
                Wsd + 0 * DPAD * DPAD, Wsd + 1 * DPAD * DPAD,
                Wsd + 2 * DPAD * DPAD, Wsd + 3 * DPAD * DPAD,
                bsd + 0 * DPAD, bsd + 1 * DPAD, bsd + 2 * DPAD, bsd + 3 * DPAD,
                lane, true, out, rowBase, &sValid[w][0], M);
}

// ============================================================================
extern "C" void kernel_launch(void* const* d_in, const int* in_sizes, int n_in,
                              void* d_out, int out_size, void* d_ws, size_t ws_size,
                              hipStream_t stream)
{
    const float* pts   = (const float*)d_in[0];
    const int*   times = (const int*)  d_in[1];
    const int*   buf   = (const int*)  d_in[2];
    const float* sf    = (const float*)d_in[3];
    const float* df    = (const float*)d_in[4];
    const float* te    = (const float*)d_in[5];
    const float* tdw   = (const float*)d_in[6];
    const float* tdb   = (const float*)d_in[7];
    const float* sdw   = (const float*)d_in[8];
    const float* sdb   = (const float*)d_in[9];
    float* out = (float*)d_out;
    int M = in_sizes[1];   // query_times count == #points

    _Float16* wts = (_Float16*)d_ws;
    float* bias = (float*)((char*)d_ws + (size_t)WTS_ELEMS * sizeof(_Float16));

    vox_prep_weights<<<(WTS_ELEMS + 255) / 256, 256, 0, stream>>>(tdw, tdb, sdw, sdb,
                                                                  wts, bias);
    int blocks = (M + 63) / 64;
    vox_fused_kernel<<<blocks, 128, 0, stream>>>(pts, times, buf, sf, df, te,
                                                 wts, bias, out, M);
}
